// WindowAttention_59407987638609
// MI455X (gfx1250) — matrix operand order repaired and checked by reading, not compiled, not run on hardware
//
#include <hip/hip_runtime.h>

typedef __attribute__((ext_vector_type(16))) _Float16 v16h;
typedef __attribute__((ext_vector_type(8)))  _Float16 v8h;
typedef __attribute__((ext_vector_type(8)))  float    v8f;

#define WMMA_F16(A, B, Cc) __builtin_amdgcn_wmma_f32_16x16x32_f16( \
    false, (A), false, (B), (short)0, (Cc), false, false)

// ---------------- problem constants ----------------
constexpr int N_TOK = 49;    // tokens per window (7x7)
constexpr int NP    = 64;    // padded rows (4 x 16 WMMA tiles)
constexpr int CH    = 128;   // channels
constexpr int HEADS = 4;
constexpr int DH    = 32;    // head dim
constexpr int NWIN  = 1024;  // mask windows

// ---------------- LDS layout (halves / floats strides) ----------------
constexpr int XS  = 136;     // Xh / Oh stride (halves)  -> 272B rows, 16B aligned
constexpr int QKS = 264;     // QK stride (halves)       -> 528B rows
constexpr int PS  = 72;      // P stride (halves)        -> 144B rows
constexpr int VS  = 72;      // Vt stride (halves)
constexpr int SS  = 68;      // S stride (floats)        -> 272B rows

// region A: Xh (P1-P2) -> S (P3) -> Oh (P4-P5)   : 4*64*68*4 = 69632 B
// region B: QK (P2-P3a) -> P (P3b-P4)            : 4*64*72*2 = 36864 B
// region C: Vt (P2-P4)                           : 4*32*72*2 = 18432 B
constexpr int OFF_A = 0;
constexpr int OFF_B = 69632;
constexpr int OFF_C = 69632 + 36864;           // 106496
constexpr int SMEM_BYTES = OFF_C + 18432;      // 124928

// ---------------- fragment helpers ----------------
// A-matrix 16x32 f16 fragment (contiguous K order, matching B fragments):
// lanes 0-15: M=lane, K = {0..15}; lanes 16-31: K = {16..31}
__device__ inline v16h load_a_frag(const _Float16* p, int stride, int ln, int lh) {
    const _Float16* r = p + ln * stride + (lh << 4);
    v8h lo = *(const v8h*)r;          // K = kb .. kb+7
    v8h hi = *(const v8h*)(r + 8);    // K = kb+8 .. kb+15
    v16h a;
#pragma unroll
    for (int i = 0; i < 8; ++i) { a[i] = lo[i]; a[i + 8] = hi[i]; }
    return a;
}

// B-matrix 32x16 f16 fragment: lane holds column N=lane%16, 16 consecutive K
// values starting at K = 16*(lane/16). Caller passes the per-lane address.
__device__ inline v16h load_row16(const _Float16* p) {
    v8h lo = *(const v8h*)p;
    v8h hi = *(const v8h*)(p + 8);
    v16h b;
#pragma unroll
    for (int i = 0; i < 8; ++i) { b[i] = lo[i]; b[i + 8] = hi[i]; }
    return b;
}

// 16 consecutive fp32 -> v16h (B fragment from global weights, L2-resident)
__device__ inline v16h cvt_row16(const float* p) {
    const float4* q = (const float4*)p;
    float4 f0 = q[0], f1 = q[1], f2 = q[2], f3 = q[3];
    v16h r;
    r[0]  = (_Float16)f0.x; r[1]  = (_Float16)f0.y; r[2]  = (_Float16)f0.z; r[3]  = (_Float16)f0.w;
    r[4]  = (_Float16)f1.x; r[5]  = (_Float16)f1.y; r[6]  = (_Float16)f1.z; r[7]  = (_Float16)f1.w;
    r[8]  = (_Float16)f2.x; r[9]  = (_Float16)f2.y; r[10] = (_Float16)f2.z; r[11] = (_Float16)f2.w;
    r[12] = (_Float16)f3.x; r[13] = (_Float16)f3.y; r[14] = (_Float16)f3.z; r[15] = (_Float16)f3.w;
    return r;
}

// ---------------- fused window attention kernel ----------------
// 1 workgroup = 1 window; 4 waves (wave32) = 4 heads in the attention phases.
__global__ __launch_bounds__(128)
void win_attn_fused(const float* __restrict__ x, const float* __restrict__ mask,
                    const float* __restrict__ qkv_w, const float* __restrict__ qkv_b,
                    const float* __restrict__ proj_w, const float* __restrict__ proj_b,
                    const float* __restrict__ bias_table, float* __restrict__ out) {
    extern __shared__ char smem[];
    _Float16* XH = (_Float16*)(smem + OFF_A);
    float*    SB = (float*)   (smem + OFF_A);
    _Float16* OH = (_Float16*)(smem + OFF_A);
    _Float16* QK = (_Float16*)(smem + OFF_B);
    _Float16* PB = (_Float16*)(smem + OFF_B);
    _Float16* VT = (_Float16*)(smem + OFF_C);

    const int tid  = threadIdx.x;
    const int lane = tid & 31;
    const int wv   = tid >> 5;        // wave id 0..3
    const int ln   = lane & 15;
    const int lh   = lane >> 4;       // half-wave select
    const int w    = blockIdx.x;      // window id

    // ---- Phase 1: load x window -> f16 LDS (padded rows zeroed) ----
    {
        const float* xw = x + (size_t)w * (N_TOK * CH);
        for (int idx = tid; idx < N_TOK * CH / 4; idx += 128) {
            int row = idx >> 5;            // 32 float4 per row
            int c4  = idx & 31;
            float4 f = ((const float4*)xw)[(size_t)row * 32 + c4];
            _Float16* d = XH + row * XS + c4 * 4;
            d[0] = (_Float16)f.x; d[1] = (_Float16)f.y;
            d[2] = (_Float16)f.z; d[3] = (_Float16)f.w;
        }
        for (int idx = tid; idx < (NP - N_TOK) * XS; idx += 128)
            XH[N_TOK * XS + idx] = (_Float16)0.f;
    }
    __syncthreads();

    // ---- Phase 2: QKV GEMM  [64x128] x [128x384] ----
    // Q,K rows stored to QK; V stored transposed into Vt (for P@V B-fragments).
    {
        const float qscale = 0.17677669529663687f;   // 1/sqrt(32), folded into Q
        for (int nt = wv * 6; nt < wv * 6 + 6; ++nt) {
            const int n0   = nt * 16;
            const int colB = n0 + ln;
            v16h Bf[4];
#pragma unroll
            for (int kk = 0; kk < 4; ++kk)
                Bf[kk] = cvt_row16(qkv_w + (size_t)colB * CH + kk * 32 + (lh << 4));
            const float bv = qkv_b[colB];
            for (int mt = 0; mt < 4; ++mt) {
                v8f acc = {};
#pragma unroll
                for (int kk = 0; kk < 4; ++kk)
                    acc = WMMA_F16(load_a_frag(XH + mt * 16 * XS + kk * 32, XS, ln, lh),
                                   Bf[kk], acc);
                if (n0 < 256) {                       // Q (cols<128) or K
                    const float sc = (n0 < 128) ? qscale : 1.f;
#pragma unroll
                    for (int r = 0; r < 8; ++r) {
                        int row = mt * 16 + r + (lh << 3);
                        QK[row * QKS + colB] = (_Float16)((acc[r] + bv) * sc);
                    }
                } else {                              // V -> transposed store
                    const int hh = (colB - 256) >> 5, jj = (colB - 256) & 31;
                    _Float16* vrow = VT + hh * (DH * VS) + jj * VS;
#pragma unroll
                    for (int r = 0; r < 8; ++r)
                        vrow[mt * 16 + r + (lh << 3)] = (_Float16)(acc[r] + bv);
                }
            }
        }
    }
    __syncthreads();

    // ---- Phase 3a: S = Q K^T + rel-pos bias + window mask (per head) ----
    {
        const int h = wv;
        float* Sh = SB + h * (NP * SS);
        const float* mrow = mask + (size_t)(w & (NWIN - 1)) * (N_TOK * N_TOK);
        for (int mt = 0; mt < 4; ++mt) {
            v16h a = load_a_frag(QK + mt * 16 * QKS + h * DH, QKS, ln, lh);
            for (int nt = 0; nt < 4; ++nt) {
                v16h b = load_row16(QK + (nt * 16 + ln) * QKS + CH + h * DH + (lh << 4));
                v8f acc = {};
                acc = WMMA_F16(a, b, acc);
                const int jj = nt * 16 + ln;
#pragma unroll
                for (int r = 0; r < 8; ++r) {
                    const int ii = mt * 16 + r + (lh << 3);
                    float v = -1e30f;
                    if (ii < N_TOK && jj < N_TOK) {
                        const int qy = ii / 7, qx = ii - qy * 7;
                        const int ky = jj / 7, kx = jj - ky * 7;
                        const int bidx = (qy - ky + 6) * 13 + (qx - kx + 6);
                        v = acc[r] + bias_table[bidx * HEADS + h] + mrow[ii * N_TOK + jj];
                    }
                    Sh[ii * SS + jj] = v;
                }
            }
        }
    }
    __syncthreads();   // all waves done reading QK -> P may overwrite region B

    // ---- Phase 3b: softmax rows -> P f16 (padded cols/rows zeroed) ----
    {
        const int h = wv;
        float*    Sh = SB + h * (NP * SS);
        _Float16* Ph = PB + h * (NP * PS);
        for (int row = lane; row < NP; row += 32) {
            _Float16* pr = Ph + row * PS;
            if (row < N_TOK) {
                float* sr = Sh + row * SS;
                float mx = -1e30f;
                for (int j = 0; j < N_TOK; ++j) mx = fmaxf(mx, sr[j]);
                float sum = 0.f;
                for (int j = 0; j < N_TOK; ++j) {
                    float e = __expf(sr[j] - mx);
                    sum += e; sr[j] = e;
                }
                float inv = 1.f / sum;
                for (int j = 0; j < N_TOK; ++j) pr[j] = (_Float16)(sr[j] * inv);
                for (int j = N_TOK; j < NP; ++j) pr[j] = (_Float16)0.f;
            } else {
                for (int j = 0; j < NP; ++j) pr[j] = (_Float16)0.f;
            }
        }
    }
    __syncthreads();   // S dead -> Oh may overwrite region A

    // ---- Phase 4: O = P V  (K=64 -> two WMMA steps) ----
    {
        const int h = wv;
        const _Float16* Ph = PB + h * (NP * PS);
        const _Float16* Vh = VT + h * (DH * VS);
        for (int mt = 0; mt < 4; ++mt) {
            v16h a0 = load_a_frag(Ph + mt * 16 * PS, PS, ln, lh);
            v16h a1 = load_a_frag(Ph + mt * 16 * PS + 32, PS, ln, lh);
            for (int nt = 0; nt < 2; ++nt) {
                const _Float16* vr = Vh + (nt * 16 + ln) * VS;
                v16h b0 = load_row16(vr + (lh << 4));
                v16h b1 = load_row16(vr + 32 + (lh << 4));
                v8f acc = {};
                acc = WMMA_F16(a0, b0, acc);
                acc = WMMA_F16(a1, b1, acc);
                const int col = h * DH + nt * 16 + ln;
#pragma unroll
                for (int r = 0; r < 8; ++r)
                    OH[(mt * 16 + r + (lh << 3)) * XS + col] = (_Float16)acc[r];
            }
        }
    }
    __syncthreads();

    // ---- Phase 5: Y = O proj_w^T + proj_b, store valid rows fp32 ----
    {
        float* ow = out + (size_t)w * (N_TOK * CH);
        for (int t = 0; t < 2; ++t) {
            const int n0   = (wv * 2 + t) * 16;
            const int colB = n0 + ln;
            v16h Bf[4];
#pragma unroll
            for (int kk = 0; kk < 4; ++kk)
                Bf[kk] = cvt_row16(proj_w + (size_t)colB * CH + kk * 32 + (lh << 4));
            const float pb = proj_b[colB];
            for (int mt = 0; mt < 4; ++mt) {
                v8f acc = {};
#pragma unroll
                for (int kk = 0; kk < 4; ++kk)
                    acc = WMMA_F16(load_a_frag(OH + mt * 16 * XS + kk * 32, XS, ln, lh),
                                   Bf[kk], acc);
#pragma unroll
                for (int r = 0; r < 8; ++r) {
                    const int ii = mt * 16 + r + (lh << 3);
                    if (ii < N_TOK) ow[(size_t)ii * CH + colB] = acc[r] + pb;
                }
            }
        }
    }
}

extern "C" void kernel_launch(void* const* d_in, const int* in_sizes, int n_in,
                              void* d_out, int out_size, void* d_ws, size_t ws_size,
                              hipStream_t stream) {
    (void)in_sizes; (void)n_in; (void)out_size; (void)d_ws; (void)ws_size;
    const float* x          = (const float*)d_in[0];
    const float* mask       = (const float*)d_in[1];
    const float* qkv_w      = (const float*)d_in[2];
    const float* qkv_b      = (const float*)d_in[3];
    const float* proj_w     = (const float*)d_in[4];
    const float* proj_b     = (const float*)d_in[5];
    const float* bias_table = (const float*)d_in[6];
    float* out = (float*)d_out;

    // 122 KB dynamic LDS (> default 64 KB static limit); opt in every call —
    // deterministic, graph-capture safe (not a stream op).
    hipFuncSetAttribute(reinterpret_cast<const void*>(win_attn_fused),
                        hipFuncAttributeMaxDynamicSharedMemorySize, SMEM_BYTES);

    win_attn_fused<<<16384, 128, SMEM_BYTES, stream>>>(
        x, mask, qkv_w, qkv_b, proj_w, proj_b, bias_table, out);
}
